// AttentionBlock_73280732004598
// MI455X (gfx1250) — matrix operand order repaired and checked by reading, not compiled, not run on hardware
//
#include <hip/hip_runtime.h>

// ---------------------------------------------------------------------------
// SAGAN attention block, MI455X (gfx1250, wave32, WMMA bf16 16x16x32 + TDM/async)
//  B=16 H=W=64 C=512 -> S=4096 queries, Sk=1024 keys, d=64, v=256
// ---------------------------------------------------------------------------

#define AS1 __attribute__((address_space(1)))
#define AS3 __attribute__((address_space(3)))

typedef __attribute__((ext_vector_type(16))) __bf16       v16bf;
typedef __attribute__((ext_vector_type(8)))  float        v8f;
typedef __attribute__((ext_vector_type(8)))  unsigned int v8u;
typedef __attribute__((ext_vector_type(4)))  unsigned int v4u;
typedef __attribute__((ext_vector_type(4)))  int          v4i;
typedef __attribute__((ext_vector_type(8)))  int          v8i;

union frag_u { v8u u; v16bf b; };

#if defined(__has_builtin)
# if __has_builtin(__builtin_amdgcn_global_load_async_to_lds_b128)
#  define HAVE_ASYNC_BUILTIN 1
# else
#  define HAVE_ASYNC_BUILTIN 0
# endif
# if __has_builtin(__builtin_amdgcn_tensor_load_to_lds)
#  define HAVE_TDM 1
# else
#  define HAVE_TDM 0
# endif
#else
# define HAVE_ASYNC_BUILTIN 0
# define HAVE_TDM 0
#endif

#if defined(__has_include)
# if __has_include(<hip/amd_detail/amd_gfx1250_TDM.h>)
#  define TDM_6ARG 1
# else
#  define TDM_6ARG 0
# endif
#else
# define TDM_6ARG 0
#endif

static __device__ __forceinline__ unsigned short f2bf(float f) {
    unsigned int u = __builtin_bit_cast(unsigned int, f);
    unsigned int r = u + 0x7fffu + ((u >> 16) & 1u);   // round-to-nearest-even
    return (unsigned short)(r >> 16);
}
static __device__ __forceinline__ float bf2f(unsigned short h) {
    unsigned int u = ((unsigned int)h) << 16;
    return __builtin_bit_cast(float, u);
}
// packed f32x2 -> bf16x2 (one v_cvt_pk_bf16_f32 when available)
static __device__ __forceinline__ unsigned int f2bf_pk(float a, float b) {
#if __has_builtin(__builtin_amdgcn_cvt_pk_bf16_f32)
    auto r = __builtin_amdgcn_cvt_pk_bf16_f32(a, b);
    return __builtin_bit_cast(unsigned int, r);
#else
    return (unsigned int)f2bf(a) | ((unsigned int)f2bf(b) << 16);
#endif
}

// --- gfx1250 async global->LDS 16-byte copy (per-lane) ----------------------
static __device__ __forceinline__ void copy16_g2l(void* l, const void* g) {
#if HAVE_ASYNC_BUILTIN
    __builtin_amdgcn_global_load_async_to_lds_b128(
        (AS1 v4i*)(size_t)g, (AS3 v4i*)l, 0, 0);
#else
    unsigned lofs = (unsigned)(size_t)(AS3 void*)l;
    asm volatile("global_load_async_to_lds_b128 %0, %1, off"
                 :: "v"(lofs), "v"(g) : "memory");
#endif
}
static __device__ __forceinline__ void wait_async0() {
#if __has_builtin(__builtin_amdgcn_s_wait_asynccnt)
    __builtin_amdgcn_s_wait_asynccnt(0);
#else
    asm volatile("s_wait_asynccnt 0" ::: "memory");
#endif
}
static __device__ __forceinline__ void wait_tensor0() {
#if __has_builtin(__builtin_amdgcn_s_wait_tensorcnt)
    __builtin_amdgcn_s_wait_tensorcnt(0);
#else
    asm volatile("s_wait_tensorcnt 0" ::: "memory");
#endif
}

#if HAVE_TDM
// TDM: linear 1-D tile of n bf16 elements, global -> LDS
static __device__ __forceinline__ void tdm_load_lin(void* lds, const void* g, int n) {
    unsigned lptr = (unsigned)(size_t)(AS3 void*)lds;
    unsigned long long ga = (unsigned long long)(size_t)g;
    v4u g0;
    g0[0] = 1u;                                   // count=1, user mode
    g0[1] = lptr;                                 // lds_addr (bytes)
    g0[2] = (unsigned)ga;                         // global_addr[31:0]
    g0[3] = (unsigned)((ga >> 32) & 0x1ffffffu) | 0x80000000u;  // addr[56:32] | type=2
    v8i g1;
    g1[0] = 0x00010000;                           // data_size=1 (2 bytes)
    g1[1] = (int)(((unsigned)n & 0xffffu) << 16); // tensor_dim0[15:0]
    g1[2] = (int)((((unsigned)n >> 16) & 0xffffu) | (1u << 16)); // dim0 hi | tensor_dim1=1
    g1[3] = (int)(((unsigned)n & 0xffffu) << 16); // tile_dim0 (n < 65536)
    g1[4] = 0;                                    // tile_dim1/2 unused
    g1[5] = n;                                    // tensor_dim0_stride lo
    g1[6] = 0;
    g1[7] = 0;
    v4i z4; z4[0] = 0; z4[1] = 0; z4[2] = 0; z4[3] = 0;
#if TDM_6ARG
    v8i z8;
    for (int i = 0; i < 8; i++) z8[i] = 0;
    __builtin_amdgcn_tensor_load_to_lds(g0, g1, z4, z4, z8, 0);
#else
    __builtin_amdgcn_tensor_load_to_lds(g0, g1, z4, z4, 0);
#endif
}
#endif

#define WMMA_BF16(A, Bm, Cm) \
    __builtin_amdgcn_wmma_f32_16x16x32_bf16(false, (A), false, (Bm), (short)0, (Cm), false, false)

// ---------------------------------------------------------------------------
// Kernel 0: weights fp32 -> bf16, TRANSPOSED layouts so LDS staging is a pure
// copy with K-pairs contiguous:  Wcat_t[384][512], Wob_t[512][256]
// ---------------------------------------------------------------------------
__global__ __launch_bounds__(256) void k_prep(const float* __restrict__ Wt,
                                              const float* __restrict__ Wp,
                                              const float* __restrict__ Wg,
                                              const float* __restrict__ Wo,
                                              unsigned short* __restrict__ Wcat_t,
                                              unsigned short* __restrict__ Wob_t) {
    int i = blockIdx.x * 256 + threadIdx.x;
    if (i < 384 * 512) {
        int c = i >> 9, k = i & 511;
        float v = (c < 64) ? Wt[k * 64 + c]
                : (c < 128) ? Wp[k * 64 + (c - 64)]
                : Wg[k * 256 + (c - 128)];
        Wcat_t[i] = f2bf(v);
    } else {
        int j = i - 384 * 512;
        if (j < 512 * 256) {
            int c = j >> 8, k = j & 255;
            Wob_t[j] = f2bf(Wo[k * 512 + c]);
        }
    }
}

// ---------------------------------------------------------------------------
// Kernel 1: projections + 2x2 maxpool.
// Block = 8 waves, 64 pixels = 16 pooled windows.  [64x512]@[512x384] via WMMA.
// ---------------------------------------------------------------------------
__global__ __launch_bounds__(256) void k_proj(const float* __restrict__ x,
                                              const unsigned short* __restrict__ Wcat_t,
                                              unsigned short* __restrict__ theta,
                                              unsigned short* __restrict__ phi,
                                              unsigned short* __restrict__ g) {
    __shared__ __align__(16) unsigned char smemraw[64 * 320 * 2];          // 40 KB
    unsigned short (*As)[32]  = (unsigned short(*)[32])smemraw;            // 64x32 bf16 (4 KB)
    unsigned short (*Bs)[32]  = (unsigned short(*)[32])(smemraw + 64*32*2);// 384x32 bf16 [col][k] (24 KB)
    unsigned short (*Os)[320] = (unsigned short(*)[320])smemraw;           // reused after GEMM

    const int tid = threadIdx.x, lane = tid & 31, wave = tid >> 5;
    const int lo = lane & 15, hi = lane >> 4;
    const int bb = blockIdx.x;
    const int b  = bb >> 6;
    const int pt = bb & 63;
    const int p0 = pt << 4;
    const int ph = p0 >> 5, pw0 = p0 & 31;
    const int h0 = ph * 2, w0 = pw0 * 2;
    const float* xb = x + (size_t)b * 4096 * 512;
    const int mt = wave & 3, nh = wave >> 2;

    v8f acc[12];
    #pragma unroll
    for (int i = 0; i < 12; i++)
        #pragma unroll
        for (int e = 0; e < 8; e++) acc[i][e] = 0.0f;

    for (int k0 = 0; k0 < 512; k0 += 32) {
        // stage A: 64 rows x 32 ch, fp32 -> bf16 (packed converts)
        for (int i = tid; i < 512; i += 256) {
            int row = i >> 3, kk4 = (i & 7) * 4;
            int q = (h0 + (row >> 5)) * 64 + w0 + (row & 31);
            float4 v = *(const float4*)(xb + (size_t)q * 512 + k0 + kk4);
            unsigned int* dst = (unsigned int*)&As[row][0];
            dst[(kk4 >> 1) + 0] = f2bf_pk(v.x, v.y);
            dst[(kk4 >> 1) + 1] = f2bf_pk(v.z, v.w);
        }
        // stage B: async copy from pre-transposed weights (pure copy, 16B/lane)
        for (int i = tid; i < 384 * 4; i += 256) {
            int c = i >> 2, ch = i & 3;
            copy16_g2l(&Bs[c][ch * 8], Wcat_t + (size_t)c * 512 + k0 + ch * 8);
        }
        wait_async0();
        __syncthreads();

        // A fragment (16x32 bf16: lane=row, K pairs split by lane-half)
        frag_u a;
        {
            const unsigned int* ar = (const unsigned int*)&As[mt * 16 + lo][0];
            #pragma unroll
            for (int j = 0; j < 4; j++) { int kk = hi * 16 + 2 * j;            a.u[j] = ar[kk >> 1]; }
            #pragma unroll
            for (int j = 4; j < 8; j++) { int kk = hi * 16 + 2 * j; a.u[j] = ar[kk >> 1]; }
        }
        #pragma unroll
        for (int idx = 0; idx < 12; idx++) {
            int nt = nh * 12 + idx;
            frag_u bf;
            const unsigned int* br = (const unsigned int*)&Bs[nt * 16 + lo][0];
            int kb = hi * 16;
            #pragma unroll
            for (int j = 0; j < 8; j++) bf.u[j] = br[(kb + 2 * j) >> 1];
            acc[idx] = WMMA_BF16(a.b, bf.b, acc[idx]);
        }
        __syncthreads();
    }

    // epilogue: theta direct, phi/g via LDS for pooling
    #pragma unroll
    for (int idx = 0; idx < 12; idx++) {
        int nt = nh * 12 + idx;
        #pragma unroll
        for (int r = 0; r < 8; r++) {
            int t = mt * 16 + r + 8 * hi;
            unsigned short v = f2bf(acc[idx][r]);
            if (nt < 4) {
                int q = (h0 + (t >> 5)) * 64 + w0 + (t & 31);
                theta[((size_t)b * 4096 + q) * 64 + nt * 16 + lo] = v;
            } else {
                Os[t][nt * 16 + lo - 64] = v;
            }
        }
    }
    __syncthreads();
    for (int i = tid; i < 16 * 320; i += 256) {
        int jj = i / 320, c = i % 320;
        float m = fmaxf(fmaxf(bf2f(Os[2 * jj][c]),      bf2f(Os[2 * jj + 1][c])),
                        fmaxf(bf2f(Os[32 + 2 * jj][c]), bf2f(Os[33 + 2 * jj][c])));
        unsigned short v = f2bf(m);
        int prow = p0 + jj;
        if (c < 64) phi[((size_t)b * 1024 + prow) * 64 + c] = v;
        else        g  [((size_t)b * 1024 + prow) * 256 + (c - 64)] = v;
    }
}

// ---------------------------------------------------------------------------
// Kernel 2: flash attention. Wave = 16 query rows, O[16x256] f32 in VGPRs.
// phi chunk staged by TDM (or async); g chunk transposed with packed b32 LDS stores.
// ---------------------------------------------------------------------------
#define KCH 64
__global__ __launch_bounds__(256) void k_attn(const unsigned short* __restrict__ theta,
                                              const unsigned short* __restrict__ phi,
                                              const unsigned short* __restrict__ g,
                                              unsigned short* __restrict__ tmp) {
    __shared__ __align__(16) unsigned short phi_s[KCH][64];   // 8 KB, [key][feature]
    __shared__ unsigned short g_s[256][KCH];                  // 32 KB, [col][key]
    __shared__ unsigned short p_s[8][16][32];                 // 8 KB per-wave P scratch

    const int tid = threadIdx.x, lane = tid & 31, wave = tid >> 5;
    const int lo = lane & 15, hi = lane >> 4;
    const int blk = blockIdx.x;
    const int b   = blk >> 5;
    const int q0  = ((blk & 31) * 8 + wave) * 16;

    frag_u a0, a1;
    {
        const unsigned int* thu =
            (const unsigned int*)(theta + ((size_t)b * 4096 + q0 + lo) * 64);
        #pragma unroll
        for (int j = 0; j < 4; j++) { int kk = hi * 16 + 2 * j;            a0.u[j] = thu[kk >> 1]; a1.u[j] = thu[(kk + 32) >> 1]; }
        #pragma unroll
        for (int j = 4; j < 8; j++) { int kk = hi * 16 + 2 * j; a0.u[j] = thu[kk >> 1]; a1.u[j] = thu[(kk + 32) >> 1]; }
    }

    float mrow[8], lrow[8];
    #pragma unroll
    for (int r = 0; r < 8; r++) { mrow[r] = -1e30f; lrow[r] = 0.0f; }
    v8f O[16];
    #pragma unroll
    for (int n = 0; n < 16; n++)
        #pragma unroll
        for (int e = 0; e < 8; e++) O[n][e] = 0.0f;

    const unsigned short* phb = phi + (size_t)b * 1024 * 64;
    const unsigned short* gb  = g   + (size_t)b * 1024 * 256;

    for (int kb = 0; kb < 1024; kb += KCH) {
        if (kb + KCH < 1024) {
            __builtin_prefetch(phb + (size_t)(kb + KCH) * 64, 0, 1);
            __builtin_prefetch(gb + (size_t)(kb + KCH) * 256, 0, 1);
        }
        // ---- stage phi chunk (contiguous 8KB): TDM or per-lane async copy
#if HAVE_TDM
        if (wave == 0) {
            tdm_load_lin(&phi_s[0][0], phb + (size_t)kb * 64, KCH * 64);
            wait_tensor0();
        }
#else
        {
            const unsigned short* src = phb + (size_t)kb * 64;
            for (int i = tid; i < KCH * 8; i += 256)           // 16B chunks
                copy16_g2l(((unsigned short*)&phi_s[0][0]) + i * 8, src + i * 8);
        }
#endif
        // ---- stage g chunk transposed: g_s[col][key], packed b32 stores
        {
            const unsigned int* gu = (const unsigned int*)(gb + (size_t)kb * 256);
            for (int i = tid; i < (KCH / 2) * 128; i += 256) {
                int key2 = (i >> 7) * 2, cp = i & 127;
                unsigned int u0 = gu[key2 * 128 + cp];
                unsigned int u1 = gu[(key2 + 1) * 128 + cp];
                unsigned int w0 = (u0 & 0xffffu) | (u1 << 16);
                unsigned int w1 = (u0 >> 16) | (u1 & 0xffff0000u);
                *(unsigned int*)&g_s[2 * cp][key2]     = w0;
                *(unsigned int*)&g_s[2 * cp + 1][key2] = w1;
            }
        }
#if !HAVE_TDM
        wait_async0();
#endif
        __syncthreads();

        #pragma unroll
        for (int kp = 0; kp < KCH / 32; kp++) {
            const int keybase = kp * 32;
            v8f s0, s1;
            {
                frag_u bk0, bk1;
                const unsigned int* pr = (const unsigned int*)&phi_s[keybase + lo][0];
                #pragma unroll
                for (int j = 0; j < 8; j++) { int kk = hi * 16 + 2 * j; bk0.u[j] = pr[kk >> 1]; bk1.u[j] = pr[(kk + 32) >> 1]; }
                v8f z;
                #pragma unroll
                for (int e = 0; e < 8; e++) z[e] = 0.0f;
                s0 = WMMA_BF16(a0.b, bk0.b, z);
                s0 = WMMA_BF16(a1.b, bk1.b, s0);
            }
            {
                frag_u bk0, bk1;
                const unsigned int* pr = (const unsigned int*)&phi_s[keybase + 16 + lo][0];
                #pragma unroll
                for (int j = 0; j < 8; j++) { int kk = hi * 16 + 2 * j; bk0.u[j] = pr[kk >> 1]; bk1.u[j] = pr[(kk + 32) >> 1]; }
                v8f z;
                #pragma unroll
                for (int e = 0; e < 8; e++) z[e] = 0.0f;
                s1 = WMMA_BF16(a0.b, bk0.b, z);
                s1 = WMMA_BF16(a1.b, bk1.b, s1);
            }
            // online softmax (row lives across a 16-lane group)
            #pragma unroll
            for (int r = 0; r < 8; r++) {
                float mx = fmaxf(s0[r], s1[r]);
                mx = fmaxf(mx, __shfl_xor(mx, 1));
                mx = fmaxf(mx, __shfl_xor(mx, 2));
                mx = fmaxf(mx, __shfl_xor(mx, 4));
                mx = fmaxf(mx, __shfl_xor(mx, 8));
                float newm = fmaxf(mrow[r], mx);
                float corr = __expf(mrow[r] - newm);
                float p0 = __expf(s0[r] - newm);
                float p1 = __expf(s1[r] - newm);
                float rs = p0 + p1;
                rs += __shfl_xor(rs, 1);
                rs += __shfl_xor(rs, 2);
                rs += __shfl_xor(rs, 4);
                rs += __shfl_xor(rs, 8);
                lrow[r] = lrow[r] * corr + rs;
                mrow[r] = newm;
                #pragma unroll
                for (int n = 0; n < 16; n++) O[n][r] *= corr;
                p_s[wave][r + 8 * hi][lo]      = f2bf(p0);
                p_s[wave][r + 8 * hi][16 + lo] = f2bf(p1);
            }
            asm volatile("s_wait_dscnt 0" ::: "memory");  // intra-wave LDS RAW
            frag_u pa;
            {
                const unsigned int* pr = (const unsigned int*)&p_s[wave][lo][0];
                #pragma unroll
                for (int j = 0; j < 4; j++) { int kk = hi * 16 + 2 * j;            pa.u[j] = pr[kk >> 1]; }
                #pragma unroll
                for (int j = 4; j < 8; j++) { int kk = hi * 16 + 2 * j; pa.u[j] = pr[kk >> 1]; }
            }
            #pragma unroll
            for (int n = 0; n < 16; n++) {
                frag_u bg;
                const unsigned int* gr = (const unsigned int*)&g_s[n * 16 + lo][0];
                #pragma unroll
                for (int j = 0; j < 8; j++) { int kk = keybase + hi * 16 + 2 * j; bg.u[j] = gr[kk >> 1]; }
                O[n] = WMMA_BF16(pa.b, bg.b, O[n]);
            }
        }
        __syncthreads();
    }

    #pragma unroll
    for (int r = 0; r < 8; r++) {
        float inv = 1.0f / lrow[r];
        #pragma unroll
        for (int n = 0; n < 16; n++) O[n][r] *= inv;
    }
    unsigned short* tb = tmp + ((size_t)b * 4096 + q0) * 256;
    #pragma unroll
    for (int n = 0; n < 16; n++)
        #pragma unroll
        for (int r = 0; r < 8; r++)
            tb[(size_t)(r + 8 * hi) * 256 + n * 16 + lo] = f2bf(O[n][r]);
}

// ---------------------------------------------------------------------------
// Kernel 3: o = tmp @ W_o ; out = gamma*o + x
// ---------------------------------------------------------------------------
__global__ __launch_bounds__(256) void k_oproj(const unsigned short* __restrict__ tmp,
                                               const unsigned short* __restrict__ Wob_t,
                                               const float* __restrict__ x,
                                               const float* __restrict__ gammap,
                                               float* __restrict__ out) {
    __shared__ __align__(16) unsigned short Ws[512][32];      // 32 KB, [col][k]
    const int tid = threadIdx.x, lane = tid & 31, wave = tid >> 5;
    const int lo = lane & 15, hi = lane >> 4;
    const int gq0 = blockIdx.x * 16;
    const int nbase = wave * 64;

    v8f acc4[4];
    #pragma unroll
    for (int n = 0; n < 4; n++)
        #pragma unroll
        for (int e = 0; e < 8; e++) acc4[n][e] = 0.0f;

    const unsigned int* tu = (const unsigned int*)(tmp + (size_t)(gq0 + lo) * 256);

    for (int k0 = 0; k0 < 256; k0 += 32) {
        // stage Ws from pre-transposed Wo: pure async copy
        for (int i = tid; i < 512 * 4; i += 256) {
            int c = i >> 2, ch = i & 3;
            copy16_g2l(&Ws[c][ch * 8], Wob_t + (size_t)c * 256 + k0 + ch * 8);
        }
        wait_async0();
        __syncthreads();
        frag_u a;
        #pragma unroll
        for (int j = 0; j < 4; j++) { int kk = k0 + hi * 16 + 2 * j;            a.u[j] = tu[kk >> 1]; }
        #pragma unroll
        for (int j = 4; j < 8; j++) { int kk = k0 + hi * 16 + 2 * j; a.u[j] = tu[kk >> 1]; }
        #pragma unroll
        for (int n = 0; n < 4; n++) {
            frag_u bf;
            const unsigned int* wr = (const unsigned int*)&Ws[nbase + n * 16 + lo][0];
            #pragma unroll
            for (int j = 0; j < 8; j++) { int kk = hi * 16 + 2 * j; bf.u[j] = wr[kk >> 1]; }
            acc4[n] = WMMA_BF16(a.b, bf.b, acc4[n]);
        }
        __syncthreads();
    }

    float gam = *gammap;
    #pragma unroll
    for (int n = 0; n < 4; n++)
        #pragma unroll
        for (int r = 0; r < 8; r++) {
            size_t off = (size_t)(gq0 + r + 8 * hi) * 512 + nbase + n * 16 + lo;
            out[off] = gam * acc4[n][r] + x[off];
        }
}

// ---------------------------------------------------------------------------
extern "C" void kernel_launch(void* const* d_in, const int* in_sizes, int n_in,
                              void* d_out, int out_size, void* d_ws, size_t ws_size,
                              hipStream_t stream) {
    const float* x     = (const float*)d_in[0];
    const float* Wt    = (const float*)d_in[1];
    const float* Wp    = (const float*)d_in[2];
    const float* Wg    = (const float*)d_in[3];
    const float* Wo    = (const float*)d_in[4];
    const float* gamma = (const float*)d_in[5];
    float* out = (float*)d_out;

    char* ws = (char*)d_ws;
    unsigned short* Wcat_t = (unsigned short*)(ws + 0);         // 384*512*2  = 393216
    unsigned short* Wob_t  = (unsigned short*)(ws + 393216);    // 512*256*2  = 262144
    unsigned short* theta  = (unsigned short*)(ws + 655360);    // 16*4096*64*2  = 8388608
    unsigned short* phi    = (unsigned short*)(ws + 9043968);   // 16*1024*64*2  = 2097152
    unsigned short* g      = (unsigned short*)(ws + 11141120);  // 16*1024*256*2 = 8388608
    unsigned short* tmp    = (unsigned short*)(ws + 19529728);  // 16*4096*256*2 = 33554432

    k_prep <<<1280, 256, 0, stream>>>(Wt, Wp, Wg, Wo, Wcat_t, Wob_t);
    k_proj <<<1024, 256, 0, stream>>>(x, Wcat_t, theta, phi, g);
    k_attn <<<512,  256, 0, stream>>>(theta, phi, g, tmp);
    k_oproj<<<4096, 256, 0, stream>>>(tmp, Wob_t, x, gamma, out);
}
